// Decoder_60232621359478
// MI455X (gfx1250) — matrix operand order repaired and checked by reading, not compiled, not run on hardware
//
#include <hip/hip_runtime.h>

typedef unsigned short ushort_t;
typedef unsigned int   uint32;
typedef __attribute__((ext_vector_type(16))) __bf16  v16bf;
typedef __attribute__((ext_vector_type(8)))  float   v8f;
typedef __attribute__((ext_vector_type(4)))  uint32  u32x4;

#define NTOK 8192   // B*T = 32*256

// ---------- helpers ----------
__device__ __forceinline__ uint32 f2bf_bits(float f) {
  uint32 u = __float_as_uint(f);
  return (u + 0x7FFFu + ((u >> 16) & 1u)) >> 16;   // RNE f32 -> bf16 bits
}

union FragAB {           // 16 bf16 = 8 dwords
  v16bf v;
  u32x4 q[2];
  uint32 u[8];
};

// ---------- weight convert: W (E,K,N) f32 + b (E,N) -> Wt (E,Npad,Kpad) bf16, transposed, bias row at k==K ----------
__global__ __launch_bounds__(256) void conv_w(const float* __restrict__ W,
                                              const float* __restrict__ b,
                                              ushort_t* __restrict__ Wt,
                                              int K, int N, int Npad, int Kpad) {
  long e = blockIdx.y;
  long idx = (long)blockIdx.x * 256 + threadIdx.x;
  if (idx >= (long)Npad * Kpad) return;
  int n = (int)(idx / Kpad);
  int k = (int)(idx % Kpad);
  float v = 0.f;
  if (n < N) {
    if (k < K)       v = W[(e * K + k) * N + n];
    else if (k == K) v = b[e * N + n];
  }
  Wt[(e * Npad + n) * Kpad + k] = (ushort_t)f2bf_bits(v);
}

// ---------- pack gating input: [z(256) | p(16) | 1 | 0..] width 288 ----------
__global__ __launch_bounds__(256) void pack_g0(const float* __restrict__ z,
                                               const float* __restrict__ p,
                                               ushort_t* __restrict__ out) {
  long idx = (long)blockIdx.x * 256 + threadIdx.x;   // NTOK*288 threads
  int t = (int)(idx / 288), k = (int)(idx % 288);
  float v = 0.f;
  if (k < 256)       v = z[(long)t * 256 + k];
  else if (k < 272)  v = p[(long)t * 16 + (k - 256)];
  else if (k == 272) v = 1.f;
  out[idx] = (ushort_t)f2bf_bits(v);
}

// ---------- pack z + bias-1 + zero tail into an 800-wide buffer (cols 256..767 filled later by GEMM epilogue) ----------
__global__ __launch_bounds__(256) void pack_zb(const float* __restrict__ z,
                                               ushort_t* __restrict__ out) {
  long idx = (long)blockIdx.x * 256 + threadIdx.x;   // NTOK*800 threads
  int t = (int)(idx / 800), k = (int)(idx % 800);
  if (k < 256)       out[idx] = (ushort_t)f2bf_bits(z[(long)t * 256 + k]);
  else if (k == 768) out[idx] = (ushort_t)f2bf_bits(1.f);
  else if (k > 768)  out[idx] = 0;
}

// ---------- pack MoE layer-0 input: [z(256) | v_hip(3) | x_curr(256) | 1 | 0..] width 544 ----------
__global__ __launch_bounds__(256) void pack_x0(const float* __restrict__ z,
                                               const float* __restrict__ vh,
                                               const float* __restrict__ xc,
                                               ushort_t* __restrict__ out) {
  long idx = (long)blockIdx.x * 256 + threadIdx.x;   // NTOK*544 threads
  int t = (int)(idx / 544), k = (int)(idx % 544);
  float v = 0.f;
  if (k < 256)       v = z[(long)t * 256 + k];
  else if (k < 259)  v = vh[(long)t * 3 + (k - 256)];
  else if (k < 515)  v = xc[(long)t * 256 + (k - 259)];
  else if (k == 515) v = 1.f;
  out[idx] = (ushort_t)f2bf_bits(v);
}

// ---------- softmax over E=8 per token ----------
__global__ __launch_bounds__(256) void softmax8(const float* __restrict__ logits,
                                                float* __restrict__ ew) {
  int t = blockIdx.x * 256 + threadIdx.x;
  if (t >= NTOK) return;
  float l[8], mx = -3.4e38f;
  #pragma unroll
  for (int e = 0; e < 8; ++e) { l[e] = logits[(long)t * 8 + e]; mx = fmaxf(mx, l[e]); }
  float s = 0.f;
  #pragma unroll
  for (int e = 0; e < 8; ++e) { l[e] = __expf(l[e] - mx); s += l[e]; }
  float r = 1.f / s;
  #pragma unroll
  for (int e = 0; e < 8; ++e) ew[(long)t * 8 + e] = l[e] * r;
}

// ---------- fused GEMM: out[t,n] = epi( sum_e ew[t,e] * (A_aug[t,:] . Wt_e[n,:]) ) ----------
// Wave tile = (MT*16) x (NT*16); workgroup = 4 waves stacked in M.
// Expert weighting folded into A via v_pk_mul_bf16; bias folded as extra K row.
// Software pipelined: B fragments ping-pong across (e,nt) iterations; next-K A loads
// are issued after the last expert's scaled copy is made.
template <int MT, int NT, int NEXP, int EPI>
__global__ __launch_bounds__(128) void gemm_tile(
    const ushort_t* __restrict__ A, int K,
    const ushort_t* __restrict__ W, long estride,
    const float* __restrict__ ew,
    ushort_t* __restrict__ outB, float* __restrict__ outF,
    int ldout, int col_off, int n_valid) {
  const int lane = threadIdx.x & 31;
  const int wave = threadIdx.x >> 5;
  const int h    = lane >> 4;    // lane half
  const int lm   = lane & 15;
  const int m0    = blockIdx.x * (4 * MT * 16) + wave * (MT * 16);
  const int tileN = blockIdx.y * (NT * 16);

  v8f acc[MT][NT] = {};

  // packed expert scales: two experts (bf16 each) per dword, per A-frag row
  constexpr int NP = (NEXP + 1) / 2;
  uint32 sc2[NP][MT];
  if constexpr (NEXP > 1) {
    #pragma unroll
    for (int pe = 0; pe < NP; ++pe)
      #pragma unroll
      for (int mt = 0; mt < MT; ++mt) {
        long r = m0 + mt * 16 + lm;
        uint32 lo = f2bf_bits(ew[r * 8 + 2 * pe]);
        uint32 hi = (2 * pe + 1 < NEXP) ? f2bf_bits(ew[r * 8 + 2 * pe + 1]) : 0u;
        sc2[pe][mt] = lo | (hi << 16);
      }
  }

  const ushort_t* arow[MT];
  #pragma unroll
  for (int mt = 0; mt < MT; ++mt)
    arow[mt] = A + (long)(m0 + mt * 16 + lm) * K;

  const ushort_t* brow[NT];
  #pragma unroll
  for (int nt = 0; nt < NT; ++nt)
    brow[nt] = W + (long)(tileN + nt * 16 + lm) * K;

  FragAB Af[MT], Ae[MT], bb[2];

  auto loadA = [&](int kb) {
    #pragma unroll
    for (int mt = 0; mt < MT; ++mt) {
      Af[mt].q[0] = *(const u32x4*)(arow[mt] + kb + h * 8);
      Af[mt].q[1] = *(const u32x4*)(arow[mt] + kb + 16 + h * 8);
    }
  };
  auto loadB = [&](FragAB& dst, int e, int nt, int kb) {
    const ushort_t* p = brow[nt] + (long)e * estride + kb;
    dst.q[0] = *(const u32x4*)(p + h * 8);
    dst.q[1] = *(const u32x4*)(p + 16 + h * 8);
  };

  constexpr int ITER = NEXP * NT;   // even -> pipelined; 1 -> simple path

  if constexpr (ITER == 1) {
    for (int kb = 0; kb < K; kb += 32) {
      loadA(kb);
      FragAB b;
      loadB(b, 0, 0, kb);
      #pragma unroll
      for (int mt = 0; mt < MT; ++mt)
        acc[mt][0] = __builtin_amdgcn_wmma_f32_16x16x32_bf16(
            false, Af[mt].v, false, b.v, (short)0, acc[mt][0], false, false);
    }
  } else {
    if constexpr (NEXP > 1) loadA(0);
    loadB(bb[0], 0, 0, 0);
    for (int kb = 0; kb < K; kb += 32) {
      const int kb_next = (kb + 32 < K) ? kb + 32 : 0;   // wrap: extra load, result unused
      if constexpr (NEXP == 1) loadA(kb);
      #pragma unroll
      for (int it = 0; it < ITER; ++it) {
        const int e  = it / NT;
        const int nt = it % NT;
        const int p  = it & 1;
        if (nt == 0) {
          if constexpr (NEXP > 1) {
            #pragma unroll
            for (int mt = 0; mt < MT; ++mt) {
              uint32 s = sc2[e / 2][mt];
              s = (e & 1) ? (s >> 16) : (s & 0xffffu);
              s |= s << 16;
              #pragma unroll
              for (int i = 0; i < 8; ++i)
                asm("v_pk_mul_bf16 %0, %1, %2"
                    : "=v"(Ae[mt].u[i]) : "v"(Af[mt].u[i]), "v"(s));
            }
            if (e == NEXP - 1) loadA(kb_next);   // Af free: Ae holds scaled copy
          }
        }
        // prefetch next B fragment (wraps into next K-step)
        const int it2 = it + 1;
        if (it2 < ITER) loadB(bb[p ^ 1], it2 / NT, it2 % NT, kb);
        else            loadB(bb[p ^ 1], 0, 0, kb_next);
        // compute: 1 B frag feeds MT WMMAs
        #pragma unroll
        for (int mt = 0; mt < MT; ++mt) {
          v16bf aop;
          if constexpr (NEXP > 1) aop = Ae[mt].v; else aop = Af[mt].v;
          acc[mt][nt] = __builtin_amdgcn_wmma_f32_16x16x32_bf16(
              false, aop, false, bb[p].v, (short)0, acc[mt][nt], false, false);
        }
      }
    }
  }

  // epilogue: C/D layout -> lane gives N (lm), vgpr vv gives M = h*8 + vv
  #pragma unroll
  for (int mt = 0; mt < MT; ++mt)
    #pragma unroll
    for (int nt = 0; nt < NT; ++nt) {
      const int n = tileN + nt * 16 + lm;
      if (n >= n_valid) continue;
      #pragma unroll
      for (int vv = 0; vv < 8; ++vv) {
        const int m = m0 + mt * 16 + h * 8 + vv;
        float val = acc[mt][nt][vv];
        if constexpr (EPI == 1) {
          val = val > 0.f ? val : (__expf(val) - 1.f);   // ELU(alpha=1)
          outB[(long)m * ldout + col_off + n] = (ushort_t)f2bf_bits(val);
        } else {
          outF[(long)m * ldout + n] = val;
        }
      }
    }
}

// ---------- host ----------
static inline int cdiv(long a, long b) { return (int)((a + b - 1) / b); }

extern "C" void kernel_launch(void* const* d_in, const int* in_sizes, int n_in,
                              void* d_out, int out_size, void* d_ws, size_t ws_size,
                              hipStream_t stream) {
  const float* z   = (const float*)d_in[0];
  const float* p   = (const float*)d_in[1];
  const float* vh  = (const float*)d_in[2];
  const float* xc  = (const float*)d_in[3];
  const float* gw0 = (const float*)d_in[4];
  const float* gb0 = (const float*)d_in[5];
  const float* gw1 = (const float*)d_in[6];
  const float* gb1 = (const float*)d_in[7];
  const float* gw2 = (const float*)d_in[8];
  const float* gb2 = (const float*)d_in[9];
  const float* w0  = (const float*)d_in[10];
  const float* b0  = (const float*)d_in[11];
  const float* w1  = (const float*)d_in[12];
  const float* b1  = (const float*)d_in[13];
  const float* w2  = (const float*)d_in[14];
  const float* b2  = (const float*)d_in[15];
  const float* wo  = (const float*)d_in[16];
  const float* bo  = (const float*)d_in[17];
  float* out = (float*)d_out;

  char* wp = (char*)d_ws;
  auto alloc = [&](size_t bytes) -> void* {
    void* r = (void*)wp;
    wp += (bytes + 255) & ~(size_t)255;
    return r;
  };
  // activation buffers (bf16)
  ushort_t* Ag0 = (ushort_t*)alloc((size_t)NTOK * 288 * 2);  // gating in
  ushort_t* Bu1 = (ushort_t*)alloc((size_t)NTOK * 800 * 2);  // g-h0 concat, reused as moe-h0 concat
  ushort_t* Bu2 = (ushort_t*)alloc((size_t)NTOK * 800 * 2);  // g-h1 concat, reused as moe-h1 concat
  ushort_t* Ax0 = (ushort_t*)alloc((size_t)NTOK * 544 * 2);  // moe layer-0 in
  ushort_t* Bu3 = (ushort_t*)alloc((size_t)NTOK * 800 * 2);  // moe-h2 concat
  float* logits = (float*)alloc((size_t)NTOK * 8 * 4);
  float* ewb    = (float*)alloc((size_t)NTOK * 8 * 4);
  // transposed+padded bf16 weights
  ushort_t* gw0t = (ushort_t*)alloc((size_t)512 * 288 * 2);
  ushort_t* gw1t = (ushort_t*)alloc((size_t)512 * 800 * 2);
  ushort_t* gw2t = (ushort_t*)alloc((size_t)16  * 800 * 2);
  ushort_t* w0t  = (ushort_t*)alloc((size_t)8 * 512 * 544 * 2);
  ushort_t* w1t  = (ushort_t*)alloc((size_t)8 * 512 * 800 * 2);
  ushort_t* w2t  = (ushort_t*)alloc((size_t)8 * 512 * 800 * 2);
  ushort_t* wot  = (ushort_t*)alloc((size_t)8 * 256 * 800 * 2);

  // --- weight conversion (transpose + bias row + pad) ---
  conv_w<<<dim3(cdiv(512L * 288, 256), 1), 256, 0, stream>>>(gw0, gb0, gw0t, 272, 512, 512, 288);
  conv_w<<<dim3(cdiv(512L * 800, 256), 1), 256, 0, stream>>>(gw1, gb1, gw1t, 768, 512, 512, 800);
  conv_w<<<dim3(cdiv(16L  * 800, 256), 1), 256, 0, stream>>>(gw2, gb2, gw2t, 768, 8,   16,  800);
  conv_w<<<dim3(cdiv(512L * 544, 256), 8), 256, 0, stream>>>(w0, b0, w0t, 515, 512, 512, 544);
  conv_w<<<dim3(cdiv(512L * 800, 256), 8), 256, 0, stream>>>(w1, b1, w1t, 768, 512, 512, 800);
  conv_w<<<dim3(cdiv(512L * 800, 256), 8), 256, 0, stream>>>(w2, b2, w2t, 768, 512, 512, 800);
  conv_w<<<dim3(cdiv(256L * 800, 256), 8), 256, 0, stream>>>(wo, bo, wot, 768, 256, 256, 800);

  // --- activation packing ---
  pack_g0<<<cdiv((long)NTOK * 288, 256), 256, 0, stream>>>(z, p, Ag0);
  pack_zb<<<cdiv((long)NTOK * 800, 256), 256, 0, stream>>>(z, Bu1);
  pack_zb<<<cdiv((long)NTOK * 800, 256), 256, 0, stream>>>(z, Bu2);
  pack_zb<<<cdiv((long)NTOK * 800, 256), 256, 0, stream>>>(z, Bu3);
  pack_x0<<<cdiv((long)NTOK * 544, 256), 256, 0, stream>>>(z, vh, xc, Ax0);

  // --- gating MLP ---
  gemm_tile<4, 2, 1, 1><<<dim3(32, 16), 128, 0, stream>>>(Ag0, 288, gw0t, 0, nullptr, Bu1, nullptr, 800, 256, 512);
  gemm_tile<4, 2, 1, 1><<<dim3(32, 16), 128, 0, stream>>>(Bu1, 800, gw1t, 0, nullptr, Bu2, nullptr, 800, 256, 512);
  gemm_tile<4, 1, 1, 0><<<dim3(32, 1),  128, 0, stream>>>(Bu2, 800, gw2t, 0, nullptr, nullptr, logits, 8, 0, 8);
  softmax8<<<cdiv(NTOK, 256), 256, 0, stream>>>(logits, ewb);

  // --- MoE expert stack ---
  gemm_tile<4, 2, 8, 1><<<dim3(32, 16), 128, 0, stream>>>(Ax0, 544, w0t, 512L * 544, ewb, Bu1, nullptr, 800, 256, 512);
  gemm_tile<4, 2, 8, 1><<<dim3(32, 16), 128, 0, stream>>>(Bu1, 800, w1t, 512L * 800, ewb, Bu2, nullptr, 800, 256, 512);
  gemm_tile<4, 2, 8, 1><<<dim3(32, 16), 128, 0, stream>>>(Bu2, 800, w2t, 512L * 800, ewb, Bu3, nullptr, 800, 256, 512);
  gemm_tile<4, 2, 8, 0><<<dim3(32, 8),  128, 0, stream>>>(Bu3, 800, wot, 256L * 800, ewb, nullptr, out, 256, 0, 256);
}
